// TinyGAT_17360257810484
// MI455X (gfx1250) — matrix operand order repaired and checked by reading, not compiled, not run on hardware
//
#include <hip/hip_runtime.h>
#include <hip/hip_bf16.h>
#include <cstddef>
#include <cstdint>

// ---------------------------------------------------------------------------
// TinyGAT on MI455X (gfx1250, wave32).
// Key algebraic fusion: s1 cancels in the row softmax, so
//   Z = (A ⊙ outer-weights) @ H / rowsum  ==  (A @ P') row-normalized,
// where P'[j,:64] = exp(s2_j)*H[j], P'[j,64] = exp(s2_j).
// Dominant cost = streaming A (256 MB @ 23.3 TB/s ~= 11 us) through
// v_wmma_f32_16x16x32_f16 with in-register int->f16 {0,1} conversion.
// ---------------------------------------------------------------------------

typedef _Float16 v16h __attribute__((ext_vector_type(16)));
typedef float    v8f  __attribute__((ext_vector_type(8)));

namespace {
constexpr int kDIN  = 768;
constexpr int kDH   = 64;
constexpr int kDOUT = 64;
constexpr int kN    = 8192;
constexpr int kNP   = 80;   // padded Z' width: 64 H cols + 1 denom + 15 zero
constexpr int kKSPLIT = 8;  // K partitions for the big GEMM (parallelism)
}

union HFrag {
  v16h v;
  _Float16 h[16];
  unsigned u[8];
};

// Convert 8 consecutive f32 -> 8 f16 (two global_load_b128 + v_cvt).
__device__ __forceinline__ void cvt8(const float* __restrict__ p, _Float16* h) {
  float4 a = *(const float4*)p;
  float4 b = *(const float4*)(p + 4);
  h[0] = (_Float16)a.x; h[1] = (_Float16)a.y; h[2] = (_Float16)a.z; h[3] = (_Float16)a.w;
  h[4] = (_Float16)b.x; h[5] = (_Float16)b.y; h[6] = (_Float16)b.z; h[7] = (_Float16)b.w;
}

// 8 consecutive {0,1} int32 -> 4 dwords of packed f16 {0.0,1.0}.
// (a0 | a1<<16) * 0x3C00 is exact for a in {0,1}: low half a0*0x3C00,
// high half a1*0x3C00 (mul wraps mod 2^32 harmlessly).
__device__ __forceinline__ void mask8(const int* __restrict__ p, unsigned* u) {
  int4 a = *(const int4*)p;
  int4 b = *(const int4*)(p + 4);
  u[0] = (unsigned)(a.x | (a.y << 16)) * 0x3C00u;
  u[1] = (unsigned)(a.z | (a.w << 16)) * 0x3C00u;
  u[2] = (unsigned)(b.x | (b.y << 16)) * 0x3C00u;
  u[3] = (unsigned)(b.z | (b.w << 16)) * 0x3C00u;
}

__device__ __forceinline__ v8f wmma16(v16h a, v16h b, v8f c) {
  // D = A(16x32 f16) * B(32x16 f16) + C(16x16 f32)
  return __builtin_amdgcn_wmma_f32_16x16x32_f16(
      /*neg_a=*/false, a, /*neg_b=*/false, b,
      /*c_mod=*/(short)0, c, /*reuse_a=*/false, /*reuse_b=*/false);
}

// ---------------------------------------------------------------------------
// K1: H(8192x64) = X(8192x768) @ W(64x768)^T  via f16 WMMA, f32 accumulate.
// One wave computes a 16x64 H tile (4 accumulator tiles), K-loop of 24 steps.
// A-frag k order chosen here (contiguous; B uses the same): lane<16 -> row=lane, K=0..15;
// lane>=16 -> row=lane-16, K=16..31.
// B-frag k order chosen here (same as A): lane<16 -> col=lane, K=0..15; lane>=16 -> K=16..31.
// W rows are contiguous in K, so B-frags are direct 64B per-lane loads.
// ---------------------------------------------------------------------------
__global__ __launch_bounds__(128) void tinygat_gemm_h(
    const float* __restrict__ X, const float* __restrict__ W,
    float* __restrict__ H) {
  const int lane = threadIdx.x & 31;
  const int wave = threadIdx.x >> 5;
  const int hs   = lane >> 4;   // half-select (0: lanes 0-15, 1: lanes 16-31)
  const int ln   = lane & 15;
  const int row0 = blockIdx.x * 64 + wave * 16;
  const float* xr = X + (size_t)(row0 + ln) * kDIN;

  v8f acc[4];
#pragma unroll
  for (int t = 0; t < 4; ++t) acc[t] = (v8f){};

  for (int k0 = 0; k0 < kDIN; k0 += 32) {
    HFrag a;
    cvt8(xr + k0 + hs * 16,     a.h);      // K run 1
    cvt8(xr + k0 + hs * 16 + 8, a.h + 8);  // K run 2
#pragma unroll
    for (int t = 0; t < 4; ++t) {
      const float* wp = W + (size_t)(t * 16 + ln) * kDIN + k0 + hs * 16;
      HFrag b;
      cvt8(wp,     b.h);
      cvt8(wp + 8, b.h + 8);
      acc[t] = wmma16(a.v, b.v, acc[t]);
    }
  }
  // C/D layout: vgpr r -> row = r + hs*8, col = ln (per 16-wide N tile).
#pragma unroll
  for (int t = 0; t < 4; ++t)
#pragma unroll
    for (int r = 0; r < 8; ++r)
      H[(size_t)(row0 + r + hs * 8) * kDH + t * 16 + ln] = acc[t][r];
}

// ---------------------------------------------------------------------------
// K2: per node j: s2_j = H[j]·a2, w_j = exp(s2_j);
// write P'^T (80 x 8192 f16): rows 0..63 = w*H col-major, row 64 = w, rest 0.
// Also zero the Z' accumulator (ws is poisoned 0xAA by the harness).
// ---------------------------------------------------------------------------
__global__ __launch_bounds__(256) void tinygat_prep(
    const float* __restrict__ H, const float* __restrict__ avec,
    _Float16* __restrict__ PT, float* __restrict__ Zp) {
  const int j = blockIdx.x * 256 + threadIdx.x;  // 0..8191
  const float* hr = H + (size_t)j * kDH;
  float s2 = 0.f;
#pragma unroll
  for (int n = 0; n < kDH; ++n) s2 += hr[n] * avec[kDH + n];  // a2 = a[64:128]
  const float w = expf(s2);
#pragma unroll
  for (int n = 0; n < kDH; ++n)
    PT[(size_t)n * kN + j] = (_Float16)(w * hr[n]);
  PT[(size_t)kDH * kN + j] = (_Float16)w;   // denominator column
#pragma unroll
  for (int n = kDH + 1; n < kNP; ++n) PT[(size_t)n * kN + j] = (_Float16)0.f;
  for (int i = j; i < kN * kNP; i += kN) Zp[i] = 0.f;
}

// ---------------------------------------------------------------------------
// K3 (dominant): Z'(8192x80) += A(8192x8192, {0,1} i32 -> f16) @ P'(8192x80).
// Wave = 32 rows x 80 cols = 2x5 accumulator tiles (80 VGPRs f32).
// Grid: 64 M-blocks x KSPLIT K-chunks; K-chunk results merged with
// global_atomic_add_f32. Per K-step: 8 b128 A loads + 10 b128 P'^T loads
// + ~24 VALU mask-converts + 10 v_wmma (VALU co-executes with XDL).
// ---------------------------------------------------------------------------
__global__ __launch_bounds__(128) void tinygat_gemm_att(
    const int* __restrict__ A, const _Float16* __restrict__ PT,
    float* __restrict__ Zp) {
  const int lane  = threadIdx.x & 31;
  const int wave  = threadIdx.x >> 5;
  const int hs    = lane >> 4;
  const int ln    = lane & 15;
  const int mbase = blockIdx.x * 128 + wave * 32;
  const int kbeg  = blockIdx.y * (kN / kKSPLIT);
  const int kend  = kbeg + (kN / kKSPLIT);

  v8f acc[2][5];
#pragma unroll
  for (int mt = 0; mt < 2; ++mt)
#pragma unroll
    for (int nt = 0; nt < 5; ++nt) acc[mt][nt] = (v8f){};

  const int* ar[2] = {A + (size_t)(mbase + ln) * kN,
                      A + (size_t)(mbase + 16 + ln) * kN};

  for (int k0 = kbeg; k0 < kend; k0 += 32) {
    HFrag am[2];
#pragma unroll
    for (int mt = 0; mt < 2; ++mt) {
      const int* p = ar[mt] + k0 + hs * 16;
      mask8(p,      am[mt].u);      // K run 1
      mask8(p + 8,  am[mt].u + 4);  // K run 2
    }
    HFrag b[5];
#pragma unroll
    for (int nt = 0; nt < 5; ++nt)
      b[nt].v = *(const v16h*)(PT + (size_t)(nt * 16 + ln) * kN + k0 + hs * 16);
#pragma unroll
    for (int mt = 0; mt < 2; ++mt)
#pragma unroll
      for (int nt = 0; nt < 5; ++nt)
        acc[mt][nt] = wmma16(am[mt].v, b[nt].v, acc[mt][nt]);
  }

#pragma unroll
  for (int mt = 0; mt < 2; ++mt)
#pragma unroll
    for (int nt = 0; nt < 5; ++nt)
#pragma unroll
      for (int r = 0; r < 8; ++r)
        atomicAdd(&Zp[(size_t)(mbase + mt * 16 + r + hs * 8) * kNP +
                      nt * 16 + ln],
                  acc[mt][nt][r]);
}

// ---------------------------------------------------------------------------
// K4: out = elu( (Z'[:, :64] / Z'[:, 64]) @ W2^T + b2 ).  33 M FMA total.
// W2 (16 KB) staged in LDS; one thread per node row.
// ---------------------------------------------------------------------------
__global__ __launch_bounds__(256) void tinygat_head(
    const float* __restrict__ Zp, const float* __restrict__ W2,
    const float* __restrict__ b2, float* __restrict__ out) {
  __shared__ float W2s[kDOUT * kDH];
  __shared__ float b2s[kDOUT];
  for (int i = threadIdx.x; i < kDOUT * kDH; i += 256) W2s[i] = W2[i];
  if (threadIdx.x < kDOUT) b2s[threadIdx.x] = b2[threadIdx.x];
  __syncthreads();

  const int i = blockIdx.x * 256 + threadIdx.x;  // node row
  const float* zr = Zp + (size_t)i * kNP;
  const float den = zr[kDH];
  const float inv = (den != 0.f) ? 1.f / den : 0.f;  // all-masked row guard
  float z[kDH];
#pragma unroll
  for (int n = 0; n < kDH; ++n) z[n] = zr[n] * inv;
#pragma unroll 4
  for (int o = 0; o < kDOUT; ++o) {
    float s = b2s[o];
#pragma unroll
    for (int n = 0; n < kDH; ++n) s += z[n] * W2s[o * kDH + n];
    out[(size_t)i * kDOUT + o] = (s > 0.f) ? s : (expf(s) - 1.f);
  }
}

// ---------------------------------------------------------------------------
extern "C" void kernel_launch(void* const* d_in, const int* in_sizes, int n_in,
                              void* d_out, int out_size, void* d_ws,
                              size_t ws_size, hipStream_t stream) {
  const float* X   = (const float*)d_in[0];  // 8192 x 768
  const int*   A   = (const int*)  d_in[1];  // 8192 x 8192 {0,1}
  const float* W   = (const float*)d_in[2];  // 64 x 768
  const float* av  = (const float*)d_in[3];  // 1 x 128
  const float* W2  = (const float*)d_in[4];  // 64 x 64
  const float* b2  = (const float*)d_in[5];  // 64
  float* out = (float*)d_out;                // 8192 x 64 f32

  // Workspace layout (~6 MB): H f32 | P'^T f16 | Z' f32
  char* ws = (char*)d_ws;
  float*    H  = (float*)ws;                                       // 2 MB
  _Float16* PT = (_Float16*)(ws + (size_t)kN * kDH * sizeof(float));
  float*    Zp = (float*)(ws + (size_t)kN * kDH * sizeof(float) +
                          (size_t)kNP * kN * sizeof(_Float16));

  // K1: H = X @ W^T (f16 WMMA). 128 blocks x 4 waves x 16 rows = 8192 rows.
  tinygat_gemm_h<<<dim3(kN / 64), dim3(128), 0, stream>>>(X, W, H);

  // K2: s2, exp, P'^T, zero Z'.
  tinygat_prep<<<dim3(kN / 256), dim3(256), 0, stream>>>(H, av, PT, Zp);

  // K3: Z' = A @ P' (f16 WMMA, K split 8-way, f32 atomic merge).
  tinygat_gemm_att<<<dim3(kN / 128, kKSPLIT), dim3(128), 0, stream>>>(A, PT, Zp);

  // K4: normalize + second layer + ELU.
  tinygat_head<<<dim3(kN / 256), dim3(256), 0, stream>>>(Zp, W2, b2, out);
}
